// HoloDistillLoss_22299470201598
// MI455X (gfx1250) — compile-verified, matrix operand order checked by reading, not run on hardware
//
#include <hip/hip_runtime.h>
#include <hip/hip_bf16.h>

// ---------------------------------------------------------------------------
// HoloDistillLoss on MI455X (gfx1250, wave32, WMMA).
// All heavy GEMMs use v_wmma_f32_16x16x32_bf16 (bf16 in, f32 accumulate),
// row-type operands only (W is pre-transposed) -> inner loop is pure
// global_load_b128 + v_wmma with dual accumulators.
// B=32, N=512, Ds=1024, Dt=2048 (fixed by the reference).
// ---------------------------------------------------------------------------

typedef __bf16 bf16;
typedef __bf16 v16bf __attribute__((ext_vector_type(16)));
typedef float  v8f   __attribute__((ext_vector_type(8)));

union Frag16 {
    v16bf v;
    uint4 q[2];
    bf16  h[16];
};

#define NEG_HUGE (-3.0e38f)

__device__ __forceinline__ bf16 f2bf(float x) {
    union { unsigned u; float f; } in; in.f = x;
    unsigned r = in.u + 0x7FFFu + ((in.u >> 16) & 1u);   // round-to-nearest-even
    union { unsigned short s; bf16 b; } out;
    out.s = (unsigned short)(r >> 16);
    return out.b;
}

// ---------------------------------------------------------------------------
// Row-type bf16 WMMA GEMM: out[m,n] = c0 + sgn * sum_k A[m,k]*B[n,k] (+bias[n])
// A: (batch, Mrows, K) row-major bf16; B: (batch, Ncols, K) row-major bf16.
// One wave computes a 16x32 output strip (two 16x16 tiles sharing the A frag):
// halves A traffic and interleaves two independent WMMA accumulator chains so
// consecutive v_wmma avoid the D->C RAW hazard NOPs.
// K % 32 == 0, Mrows % 16 == 0, Ncols % 32 == 0.
// A-frag layout (16-bit A 16x32): lane&15 = row, lane<16 -> K 0..7/16..23,
// lane>=16 -> K 8..15/24..31; each 8-half group is K-contiguous => 2x b128.
// ---------------------------------------------------------------------------
template <bool HASBIAS, bool OUTF, bool OUTH>
__global__ void __launch_bounds__(32) wmma_gemm_rt(
    const bf16* __restrict__ A, const bf16* __restrict__ Bm,
    const float* __restrict__ bias,
    float* __restrict__ outF, bf16* __restrict__ outH,
    int K, int Mrows, int Ncols,
    long long strideA, long long strideB, long long strideOut,
    int ldA, int ldB, int ldOut, float c0, float sgn)
{
    const int tilesN2 = Ncols >> 5;          // 16x32 strips along N
    const int tilesM  = Mrows >> 4;
    const int tpb     = tilesM * tilesN2;
    int blk = blockIdx.x;
    int b   = blk / tpb;
    int r   = blk - b * tpb;
    int tm  = r / tilesN2;
    int tn2 = r - tm * tilesN2;

    int lane   = threadIdx.x;
    int lane15 = lane & 15;
    int khalf  = (lane < 16) ? 0 : 8;

    const bf16* Ab = A  + (long long)b * strideA;
    const bf16* Bb = Bm + (long long)b * strideB;

    int rowA = tm * 16 + lane15;
    int col0 = tn2 * 32 + lane15;
    int col1 = col0 + 16;

    v8f acc0 = {}, acc1 = {};
    if (HASBIAS) {
        float c0v = bias[col0], c1v = bias[col1];
#pragma unroll
        for (int i = 0; i < 8; ++i) { acc0[i] = c0v; acc1[i] = c1v; }
    }

    const bf16* aRow  = Ab + (long long)rowA * ldA + khalf;
    const bf16* bRow0 = Bb + (long long)col0 * ldB + khalf;
    const bf16* bRow1 = Bb + (long long)col1 * ldB + khalf;

    for (int kc = 0; kc < K; kc += 32) {
        Frag16 a, b0, b1;
        a.q[0]  = *(const uint4*)(aRow + kc);
        a.q[1]  = *(const uint4*)(aRow + kc + 16);
        b0.q[0] = *(const uint4*)(bRow0 + kc);
        b0.q[1] = *(const uint4*)(bRow0 + kc + 16);
        b1.q[0] = *(const uint4*)(bRow1 + kc);
        b1.q[1] = *(const uint4*)(bRow1 + kc + 16);
        __builtin_prefetch(aRow + kc + 128, 0, 1);   // global_prefetch_b8
        acc0 = __builtin_amdgcn_wmma_f32_16x16x32_bf16(
            false, a.v, false, b0.v, (short)0, acc0, false, false);
        acc1 = __builtin_amdgcn_wmma_f32_16x16x32_bf16(
            false, a.v, false, b1.v, (short)0, acc1, false, false);
    }

    // C/D layout: VGPR i -> row tm*16 + i (+8 for lanes 16..31), col = lane&15.
    int rbase = tm * 16 + ((lane < 16) ? 0 : 8);
    long long ob = (long long)b * strideOut;
#pragma unroll
    for (int i = 0; i < 8; ++i) {
        float v0 = c0 + sgn * acc0[i];
        float v1 = c0 + sgn * acc1[i];
        long long idx0 = ob + (long long)(rbase + i) * ldOut + col0;
        long long idx1 = idx0 + 16;
        if (OUTF) { outF[idx0] = v0;       outF[idx1] = v1;       }
        if (OUTH) { outH[idx0] = f2bf(v0); outH[idx1] = f2bf(v1); }
    }
}

// ---------------------------------------------------------------------------
// Fused structure-loss kernel: mapped = T * Gn^T via WMMA (16x32 strip), then
// partial += (Cs - mapped)^2 * mu_row * mu_col, wave-reduced via shfl_xor.
// ---------------------------------------------------------------------------
__global__ void __launch_bounds__(32) struct_loss_kernel(
    const bf16* __restrict__ T, const bf16* __restrict__ Gn,
    const float* __restrict__ Cs, const float* __restrict__ mu,
    float* __restrict__ partials, int Nn)
{
    const int tilesN2 = Nn >> 5;
    const int tilesM  = Nn >> 4;
    const int tpb     = tilesM * tilesN2;
    int blk = blockIdx.x;
    int b   = blk / tpb;
    int r   = blk - b * tpb;
    int tm  = r / tilesN2;
    int tn2 = r - tm * tilesN2;

    int lane   = threadIdx.x;
    int lane15 = lane & 15;
    int khalf  = (lane < 16) ? 0 : 8;

    long long nn = (long long)Nn * Nn;
    const bf16*  Tb = T  + (long long)b * nn;
    const bf16*  Gb = Gn + (long long)b * nn;
    const float* Cb = Cs + (long long)b * nn;
    const float* mb = mu + (long long)b * Nn;

    int col0 = tn2 * 32 + lane15;
    int col1 = col0 + 16;

    v8f acc0 = {}, acc1 = {};
    const bf16* aRow  = Tb + (long long)(tm * 16 + lane15) * Nn + khalf;
    const bf16* bRow0 = Gb + (long long)col0 * Nn + khalf;
    const bf16* bRow1 = Gb + (long long)col1 * Nn + khalf;
    for (int kc = 0; kc < Nn; kc += 32) {
        Frag16 a, b0, b1;
        a.q[0]  = *(const uint4*)(aRow + kc);
        a.q[1]  = *(const uint4*)(aRow + kc + 16);
        b0.q[0] = *(const uint4*)(bRow0 + kc);
        b0.q[1] = *(const uint4*)(bRow0 + kc + 16);
        b1.q[0] = *(const uint4*)(bRow1 + kc);
        b1.q[1] = *(const uint4*)(bRow1 + kc + 16);
        acc0 = __builtin_amdgcn_wmma_f32_16x16x32_bf16(
            false, a.v, false, b0.v, (short)0, acc0, false, false);
        acc1 = __builtin_amdgcn_wmma_f32_16x16x32_bf16(
            false, a.v, false, b1.v, (short)0, acc1, false, false);
    }

    int rbase = tm * 16 + ((lane < 16) ? 0 : 8);
    float mc0 = mb[col0], mc1 = mb[col1];
    float s = 0.0f;
#pragma unroll
    for (int i = 0; i < 8; ++i) {
        int row = rbase + i;
        float mr = mb[row];
        float d0 = Cb[(long long)row * Nn + col0] - acc0[i];
        float d1 = Cb[(long long)row * Nn + col1] - acc1[i];
        s += d0 * d0 * mr * mc0 + d1 * d1 * mr * mc1;
    }
    for (int off2 = 16; off2 > 0; off2 >>= 1)
        s += __shfl_xor(s, off2, 32);
    if (lane == 0) partials[blockIdx.x] = s;
}

// ---------------------------------------------------------------------------
// LDS-tiled transpose with f32 -> bf16 convert: dst[c*R + r] = src[r*C + c].
// R, C multiples of 32. Grid = (R/32)*(C/32) blocks of 256 threads.
// ---------------------------------------------------------------------------
__global__ void __launch_bounds__(256) transpose_f32_bf16(
    const float* __restrict__ src, bf16* __restrict__ dst, int R, int C)
{
    __shared__ float tile[32][33];
    int tilesC = C >> 5;
    int blk = blockIdx.x;
    int trx = blk / tilesC;
    int tcx = blk - trx * tilesC;
    int tx = threadIdx.x & 31;
    int ty = threadIdx.x >> 5;      // 0..7
    for (int i = ty; i < 32; i += 8)
        tile[i][tx] = src[(long long)(trx * 32 + i) * C + tcx * 32 + tx];
    __syncthreads();
    for (int i = ty; i < 32; i += 8)
        dst[(long long)(tcx * 32 + i) * R + trx * 32 + tx] = f2bf(tile[tx][i]);
}

// ---------------------------------------------------------------------------
// Row L2-norm + f32->bf16 convert. Writes optional raw and/or normalized rows.
// ---------------------------------------------------------------------------
__global__ void __launch_bounds__(256) normconv_kernel(
    const float* __restrict__ src, int D,
    bf16* __restrict__ outRaw, bf16* __restrict__ outNorm)
{
    __shared__ float red[256];
    long long r = blockIdx.x;
    int t = threadIdx.x;
    const float* row = src + r * D;
    float ss = 0.0f;
    for (int k = t; k < D; k += 256) { float x = row[k]; ss += x * x; }
    red[t] = ss; __syncthreads();
    for (int s2 = 128; s2 > 0; s2 >>= 1) {
        if (t < s2) red[t] += red[t + s2];
        __syncthreads();
    }
    float inv = 1.0f / fmaxf(sqrtf(red[0]), 1e-12f);
    for (int k = t; k < D; k += 256) {
        float x = row[k];
        if (outRaw)  outRaw[r * D + k]  = f2bf(x);
        if (outNorm) outNorm[r * D + k] = f2bf(x * inv);
    }
}

// ---------------------------------------------------------------------------
// Saliency: softmax over token L2 norms. One block per batch, Nn threads.
// ---------------------------------------------------------------------------
__global__ void __launch_bounds__(512) saliency_kernel(
    const float* __restrict__ h, int D, float* __restrict__ mu, int Nn)
{
    __shared__ float red[512];
    int b = blockIdx.x, t = threadIdx.x;
    const float* row = h + ((long long)b * Nn + t) * D;
    float ss = 0.0f;
    for (int k = 0; k < D; ++k) { float x = row[k]; ss += x * x; }
    float nrm = sqrtf(ss);
    red[t] = nrm; __syncthreads();
    for (int s2 = 256; s2 > 0; s2 >>= 1) {
        if (t < s2) red[t] = fmaxf(red[t], red[t + s2]);
        __syncthreads();
    }
    float mx = red[0]; __syncthreads();
    float e = __expf(nrm - mx);
    red[t] = e; __syncthreads();
    for (int s2 = 256; s2 > 0; s2 >>= 1) {
        if (t < s2) red[t] += red[t + s2];
        __syncthreads();
    }
    mu[(long long)b * Nn + t] = e / red[0];
}

// ---------------------------------------------------------------------------
// Log-domain Sinkhorn, 20 iterations. One block per batch; u/v in LDS.
// Online logsumexp per row (u) / column (v) over M (M is L2-resident: 32 MB).
// ---------------------------------------------------------------------------
__global__ void __launch_bounds__(512) sinkhorn_kernel(
    const float* __restrict__ Mm, const float* __restrict__ mu,
    const float* __restrict__ nu,
    float* __restrict__ uOut, float* __restrict__ vOut, int Nn)
{
    __shared__ float su[512], sv[512], slmu[512], slnu[512];
    int b = blockIdx.x, t = threadIdx.x;
    const float* Mb = Mm + (long long)b * Nn * Nn;
    slmu[t] = __logf(mu[(long long)b * Nn + t] + 1e-8f);
    slnu[t] = __logf(nu[(long long)b * Nn + t] + 1e-8f);
    su[t] = 0.0f; sv[t] = 0.0f;
    __syncthreads();

    const float invEps = 10.0f;  // 1/OT_EPS
    for (int it = 0; it < 20; ++it) {
        {   // u_i = log_mu_i - LSE_j( -M_ij/eps + v_j )
            const float* Mrow = Mb + (long long)t * Nn;
            float mx = NEG_HUGE, s = 0.0f;
            for (int j = 0; j < Nn; ++j) {
                float x = -Mrow[j] * invEps + sv[j];
                if (x > mx) { s = s * __expf(mx - x) + 1.0f; mx = x; }
                else        { s += __expf(x - mx); }
            }
            su[t] = slmu[t] - (mx + __logf(s));
        }
        __syncthreads();
        {   // v_j = log_nu_j - LSE_i( -M_ij/eps + u_i )
            float mx = NEG_HUGE, s = 0.0f;
            for (int i = 0; i < Nn; ++i) {
                float x = -Mb[(long long)i * Nn + t] * invEps + su[i];
                if (x > mx) { s = s * __expf(mx - x) + 1.0f; mx = x; }
                else        { s += __expf(x - mx); }
            }
            sv[t] = slnu[t] - (mx + __logf(s));
        }
        __syncthreads();
    }
    uOut[(long long)b * Nn + t] = su[t];
    vOut[(long long)b * Nn + t] = sv[t];
}

// ---------------------------------------------------------------------------
// Per-row Gamma sums: rowsum_i = sum_j Gamma_ij ; featrows_i = sum_j M*Gamma.
// ---------------------------------------------------------------------------
__global__ void __launch_bounds__(256) gamma_row_kernel(
    const float* __restrict__ Mm, const float* __restrict__ u,
    const float* __restrict__ v,
    float* __restrict__ rowsum, float* __restrict__ featrows, int Nn)
{
    __shared__ float rg[256], rm[256];
    long long r = blockIdx.x;            // b*N + i
    int b = (int)(r / Nn);
    int t = threadIdx.x;
    const float* Mrow = Mm + r * Nn;
    const float* vb   = v + (long long)b * Nn;
    float ui = u[r];
    float sg = 0.0f, sm = 0.0f;
    for (int j = t; j < Nn; j += 256) {
        float m = Mrow[j];
        float g = __expf(ui - m * 10.0f + vb[j]);
        sg += g; sm += m * g;
    }
    rg[t] = sg; rm[t] = sm; __syncthreads();
    for (int s2 = 128; s2 > 0; s2 >>= 1) {
        if (t < s2) { rg[t] += rg[t + s2]; rm[t] += rm[t + s2]; }
        __syncthreads();
    }
    if (t == 0) { rowsum[r] = rg[0]; featrows[r] = rm[0]; }
}

// Gamma_norm -> bf16 (elementwise).
__global__ void __launch_bounds__(256) gn_write_kernel(
    const float* __restrict__ Mm, const float* __restrict__ u,
    const float* __restrict__ v, const float* __restrict__ rowsum,
    bf16* __restrict__ Gn, int Nn)
{
    long long idx = (long long)blockIdx.x * 256 + threadIdx.x;
    long long r = idx / Nn;              // b*N + i
    int j = (int)(idx - r * Nn);
    int b = (int)(r / Nn);
    float g = __expf(u[r] - Mm[idx] * 10.0f + v[(long long)b * Nn + j]);
    Gn[idx] = f2bf(g / (rowsum[r] + 1e-8f));
}

// ---------------------------------------------------------------------------
// Matryoshka contrastive (32x32 scores, dims {256,512,1024}); tiny, one block.
// ---------------------------------------------------------------------------
__global__ void __launch_bounds__(1024) contrastive_kernel(
    const float* __restrict__ q, const float* __restrict__ p,
    float* __restrict__ outSlot)
{
    __shared__ float ssq[3][32], ssp[3][32];
    __shared__ float sc[32][33];
    __shared__ float rowpart[32];
    __shared__ float accum;
    int t = threadIdx.x;
    const int dims[3] = {256, 512, 1024};
    if (t < 32) {
        const float* row = q + t * 1024;
        float s = 0.0f; int di = 0;
        for (int k = 0; k < 1024; ++k) {
            s += row[k] * row[k];
            if (k == 255 || k == 511 || k == 1023) ssq[di++][t] = s;
        }
    } else if (t < 64) {
        int i = t - 32;
        const float* row = p + i * 1024;
        float s = 0.0f; int di = 0;
        for (int k = 0; k < 1024; ++k) {
            s += row[k] * row[k];
            if (k == 255 || k == 511 || k == 1023) ssp[di++][i] = s;
        }
    }
    if (t == 0) accum = 0.0f;
    __syncthreads();

    int i = t >> 5, j = t & 31;
    for (int di = 0; di < 3; ++di) {
        int d = dims[di];
        const float* qi = q + i * 1024;
        const float* pj = p + j * 1024;
        float dot = 0.0f;
        for (int k = 0; k < d; ++k) dot += qi[k] * pj[k];
        float nq = fmaxf(sqrtf(ssq[di][i]), 1e-12f);
        float np = fmaxf(sqrtf(ssp[di][j]), 1e-12f);
        sc[i][j] = dot / (nq * np) * (1.0f / 0.07f);
        __syncthreads();
        if (t < 32) {
            float mx = NEG_HUGE;
            for (int c = 0; c < 32; ++c) mx = fmaxf(mx, sc[t][c]);
            float s = 0.0f;
            for (int c = 0; c < 32; ++c) s += __expf(sc[t][c] - mx);
            rowpart[t] = (mx + __logf(s)) - sc[t][t];
        }
        __syncthreads();
        if (t == 0) {
            float s = 0.0f;
            for (int c = 0; c < 32; ++c) s += rowpart[c];
            accum += s * (1.0f / 32.0f);   // weight 1.0 per matryoshka dim
        }
        __syncthreads();
    }
    if (t == 0) outSlot[0] = accum;
}

// Deterministic fixed-order reduction: dst[0] = scale * sum(src[0..n)).
__global__ void __launch_bounds__(256) reduce_kernel(
    const float* __restrict__ src, int n, float scale, float* __restrict__ dst)
{
    __shared__ float red[256];
    int t = threadIdx.x;
    float s = 0.0f;
    for (int k = t; k < n; k += 256) s += src[k];
    red[t] = s; __syncthreads();
    for (int s2 = 128; s2 > 0; s2 >>= 1) {
        if (t < s2) red[t] += red[t + s2];
        __syncthreads();
    }
    if (t == 0) dst[0] = scale * red[0];
}

__global__ void finalize_kernel(const float* __restrict__ scal,
                                float* __restrict__ out)
{
    if (threadIdx.x == 0 && blockIdx.x == 0) {
        const float ALPHA = 0.5f;
        float lossq = (1.0f - ALPHA) * scal[1] + ALPHA * scal[2];
        float lossp = (1.0f - ALPHA) * scal[3] + ALPHA * scal[4];
        out[0] = scal[0] + 1.0f * (0.5f * (lossq + lossp));  // KD_WEIGHT=1
    }
}

// ---------------------------------------------------------------------------
// Host-side orchestration.
// ---------------------------------------------------------------------------
extern "C" void kernel_launch(void* const* d_in, const int* in_sizes, int n_in,
                              void* d_out, int out_size, void* d_ws, size_t ws_size,
                              hipStream_t stream)
{
    (void)in_sizes; (void)n_in; (void)out_size; (void)ws_size;
    const int Bb = 32, Nn = 512, Ds = 1024, Dt = 2048;
    const long long BN  = (long long)Bb * Nn;      // 16384
    const long long BNN = BN * Nn;                 // 8.39M

    const float* s_q_reps = (const float*)d_in[0];
    const float* s_p_reps = (const float*)d_in[1];
    const float* W_proj   = (const float*)d_in[6];
    const float* b_proj   = (const float*)d_in[7];

    char* ws = (char*)d_ws;
    size_t off = 0;
    auto alloc = [&](size_t bytes) -> size_t {
        size_t o = off;
        off += (bytes + 255) & ~(size_t)255;
        return o;
    };

    float* scal     = (float*)(ws + alloc(64 * sizeof(float)));
    float* mu       = (float*)(ws + alloc(BN * sizeof(float)));
    float* nu       = (float*)(ws + alloc(BN * sizeof(float)));
    float* uu       = (float*)(ws + alloc(BN * sizeof(float)));
    float* vv       = (float*)(ws + alloc(BN * sizeof(float)));
    float* rowsum   = (float*)(ws + alloc(BN * sizeof(float)));
    float* featrows = (float*)(ws + alloc(BN * sizeof(float)));
    float* structp  = (float*)(ws + alloc((size_t)Bb * 512 * sizeof(float)));
    bf16*  zsraw    = (bf16*)(ws + alloc((size_t)BN * Ds * sizeof(bf16)));
    bf16*  zsn      = (bf16*)(ws + alloc((size_t)BN * Ds * sizeof(bf16)));
    bf16*  ztn      = (bf16*)(ws + alloc((size_t)BN * Dt * sizeof(bf16)));
    bf16*  Wt       = (bf16*)(ws + alloc((size_t)Ds * Dt * sizeof(bf16)));  // (Dt,Ds)
    float* proj     = (float*)(ws + alloc((size_t)BN * Dt * sizeof(float)));
    bf16*  projn    = (bf16*)(ws + alloc((size_t)BN * Dt * sizeof(bf16)));
    float* Cs       = (float*)(ws + alloc((size_t)BNN * sizeof(float)));
    float* Mmat     = (float*)(ws + alloc((size_t)BNN * sizeof(float)));
    bf16*  Ctbf     = (bf16*)(ws + alloc((size_t)BNN * sizeof(bf16)));
    bf16*  Gnbf     = (bf16*)(ws + alloc((size_t)BNN * sizeof(bf16)));
    bf16*  Tbf      = (bf16*)(ws + alloc((size_t)BNN * sizeof(bf16)));

    // 16x32 strips per batch image: (N/16)*(N/32) = 32*16 = 512
    const int stripsNN = (Nn / 16) * (Nn / 32);

    // W (Ds x Dt) -> Wt (Dt x Ds) bf16, so the projection GEMM is row-type.
    transpose_f32_bf16<<<(Ds / 32) * (Dt / 32), 256, 0, stream>>>(W_proj, Wt, Ds, Dt);
    // Contrastive term -> scal[0]
    contrastive_kernel<<<1, 1024, 0, stream>>>(s_q_reps, s_p_reps, scal + 0);

    for (int side = 0; side < 2; ++side) {
        const float* zs = (const float*)d_in[side == 0 ? 2 : 3];  // s_{q,p}_states
        const float* zt = (const float*)d_in[side == 0 ? 4 : 5];  // t_{q,p}_states
        float* featSlot   = scal + (side == 0 ? 1 : 3);
        float* structSlot = scal + (side == 0 ? 2 : 4);

        normconv_kernel<<<(int)BN, 256, 0, stream>>>(zs, Ds, zsraw, zsn);
        normconv_kernel<<<(int)BN, 256, 0, stream>>>(zt, Dt, nullptr, ztn);
        saliency_kernel<<<Bb, Nn, 0, stream>>>(zs, Ds, mu, Nn);
        saliency_kernel<<<Bb, Nn, 0, stream>>>(zt, Dt, nu, Nn);

        // C_s = 1 - zsn·zsnᵀ   (f32)
        wmma_gemm_rt<false, true, false><<<Bb * stripsNN, 32, 0, stream>>>(
            zsn, zsn, nullptr, Cs, nullptr,
            Ds, Nn, Nn,
            (long long)Nn * Ds, (long long)Nn * Ds, (long long)Nn * Nn,
            Ds, Ds, Nn, 1.0f, -1.0f);
        // C_t = 1 - ztn·ztnᵀ   (bf16, only consumed inside Γ·C_t·Γᵀ)
        wmma_gemm_rt<false, false, true><<<Bb * stripsNN, 32, 0, stream>>>(
            ztn, ztn, nullptr, nullptr, Ctbf,
            Dt, Nn, Nn,
            (long long)Nn * Dt, (long long)Nn * Dt, (long long)Nn * Nn,
            Dt, Dt, Nn, 1.0f, -1.0f);
        // proj = zs·W + b   (flattened (B*N)x2048, B = Wt row-type)
        wmma_gemm_rt<true, true, false><<<(int)((BN / 16) * (Dt / 32)), 32, 0, stream>>>(
            zsraw, Wt, b_proj, proj, nullptr,
            Ds, (int)BN, Dt,
            0, 0, 0,
            Ds, Ds, Dt, 0.0f, 1.0f);
        normconv_kernel<<<(int)BN, 256, 0, stream>>>(proj, Dt, nullptr, projn);
        // M = 1 - projn·ztnᵀ   (f32, feeds Sinkhorn)
        wmma_gemm_rt<false, true, false><<<Bb * stripsNN, 32, 0, stream>>>(
            projn, ztn, nullptr, Mmat, nullptr,
            Dt, Nn, Nn,
            (long long)Nn * Dt, (long long)Nn * Dt, (long long)Nn * Nn,
            Dt, Dt, Nn, 1.0f, -1.0f);

        sinkhorn_kernel<<<Bb, Nn, 0, stream>>>(Mmat, mu, nu, uu, vv, Nn);
        gamma_row_kernel<<<(int)BN, 256, 0, stream>>>(Mmat, uu, vv, rowsum, featrows, Nn);
        reduce_kernel<<<1, 256, 0, stream>>>(featrows, (int)BN, 1.0f / Bb, featSlot);
        gn_write_kernel<<<(int)(BNN / 256), 256, 0, stream>>>(Mmat, uu, vv, rowsum, Gnbf, Nn);

        // T = Γn·C_t  (C_t symmetric => row-type B is exact; bf16 out)
        wmma_gemm_rt<false, false, true><<<Bb * stripsNN, 32, 0, stream>>>(
            Gnbf, Ctbf, nullptr, nullptr, Tbf,
            Nn, Nn, Nn,
            (long long)Nn * Nn, (long long)Nn * Nn, (long long)Nn * Nn,
            Nn, Nn, Nn, 0.0f, 1.0f);
        // struct partials: (Cs - T·Γnᵀ)² · μμᵀ, fused WMMA + weighted diff
        struct_loss_kernel<<<Bb * stripsNN, 32, 0, stream>>>(
            Tbf, Gnbf, Cs, mu, structp, Nn);
        reduce_kernel<<<1, 256, 0, stream>>>(structp, Bb * stripsNN, 1.0f / Bb, structSlot);
    }

    finalize_kernel<<<1, 32, 0, stream>>>(scal, (float*)d_out);
}
